// TransformerLayer_31413390802962
// MI455X (gfx1250) — matrix operand order repaired and checked by reading, not compiled, not run on hardware
//
#include <hip/hip_runtime.h>
#include <hip/hip_bf16.h>

// ---------------------------------------------------------------------------
// Transformer encoder layer for MI455X (gfx1250, wave32, WMMA).
// All heavy GEMMs use v_wmma_f32_16x16x32_bf16 (bf16 inputs, f32 accum).
// Tile staging uses GLOBAL_LOAD_ASYNC_TO_LDS_B128 (ASYNCcnt) when available,
// with double-buffered LDS in the GEMM so DMA overlaps WMMA.
// Shapes fixed by the reference: B=2, S=2048, D=1024, H=16, Dk=64, Dff=4096.
// ---------------------------------------------------------------------------

typedef __bf16 bf16_t;
typedef __attribute__((ext_vector_type(16))) __bf16 v16bf;
typedef __attribute__((ext_vector_type(8)))  __bf16 v8bf;
typedef __attribute__((ext_vector_type(8)))  float  v8f;
typedef int v4i __attribute__((vector_size(16)));

#define BATCH 2
#define SEQ   2048
#define DMODEL 1024
#define NHEADS 16
#define DK    64
#define DFF   4096
#define ROWS  (BATCH * SEQ)          // 4096 tokens

#define AS1 __attribute__((address_space(1)))
#define AS3 __attribute__((address_space(3)))

#if defined(__gfx1250__) && __has_builtin(__builtin_amdgcn_global_load_async_to_lds_b128) && __has_builtin(__builtin_amdgcn_s_wait_asynccnt)
#define USE_ASYNC_LDS 1
#else
#define USE_ASYNC_LDS 0
#endif

// 16-byte global -> LDS copy (async DMA on gfx1250 when available).
__device__ __forceinline__ void g2l16(bf16_t* lds, const bf16_t* g) {
#if USE_ASYNC_LDS
  __builtin_amdgcn_global_load_async_to_lds_b128(
      (AS1 v4i*)(AS1 void*)(void*)g, (AS3 v4i*)(AS3 void*)lds, 0, 0);
#else
  *(v8bf*)lds = *(const v8bf*)g;
#endif
}

__device__ __forceinline__ void async_fence() {
#if USE_ASYNC_LDS
  __builtin_amdgcn_s_wait_asynccnt(0);
#endif
}

// -------------------------------- fragment loaders -------------------------
// A-matrix 16x32 bf16 (ISA 7.12.2): lane = M row (0..15, dup in both halves),
// lane<16 holds K = 0..7 (V0..3) and 16..23 (V4..7); lane>=16 holds K=8..15, 24..31.
__device__ __forceinline__ v16bf load_fragA(const bf16_t* base, int stride, int lane) {
  int r = lane & 15, kh = (lane >> 4) & 1;
  const bf16_t* p = base + r * stride + kh * 8;
  v8bf lo = *(const v8bf*)p;          // K = kh*8 .. kh*8+7
  v8bf hi = *(const v8bf*)(p + 16);   // K = 16 + kh*8 .. +7
  v16bf f;
#pragma unroll
  for (int i = 0; i < 8; ++i) { f[i] = lo[i]; f[i + 8] = hi[i]; }
  return f;
}

// B-matrix 32x16 bf16: lane = N column; lane<16 holds K=0..7, 16..23,
// lane>=16 holds K=8..15, 24..31. `base` points at an N-major [16 x >=32K] region.
__device__ __forceinline__ v16bf load_fragB(const bf16_t* base, int stride, int lane) {
  int r = lane & 15, kh = (lane >> 4) & 1;
  const bf16_t* p = base + r * stride + kh * 8;
  v8bf lo = *(const v8bf*)p;
  v8bf hi = *(const v8bf*)(p + 16);
  v16bf f;
#pragma unroll
  for (int i = 0; i < 8; ++i) { f[i] = lo[i]; f[i + 8] = hi[i]; }
  return f;
}

__device__ __forceinline__ v8f wmma_bf16(v16bf a, v16bf b, v8f c) {
  return __builtin_amdgcn_wmma_f32_16x16x32_bf16(false, a, false, b,
                                                 (short)0, c, false, false);
}

// -------------------------------- prep kernels -----------------------------
// Wt[n*K + k] = (bf16) W[k*N + n]
__global__ __launch_bounds__(256) void cvt_transpose_kernel(
    const float* __restrict__ W, bf16_t* __restrict__ Wt, int K, int N) {
  long t = (long)blockIdx.x * 256 + threadIdx.x;
  long n = t / K, k = t % K;
  Wt[t] = (bf16_t)W[k * (long)N + n];
}

// LayerNorm over D=1024, one block per row, bf16 output for GEMM consumption.
__global__ __launch_bounds__(256) void layernorm_kernel(
    const float* __restrict__ x, const float* __restrict__ g,
    const float* __restrict__ beta, bf16_t* __restrict__ out) {
  __shared__ float red[256];
  int row = blockIdx.x, tid = threadIdx.x;
  const float* xr = x + (long)row * DMODEL;
  float v[4]; float s = 0.f;
#pragma unroll
  for (int i = 0; i < 4; ++i) { v[i] = xr[tid + i * 256]; s += v[i]; }
  red[tid] = s; __syncthreads();
  for (int off = 128; off; off >>= 1) { if (tid < off) red[tid] += red[tid + off]; __syncthreads(); }
  float mu = red[0] * (1.f / DMODEL);
  __syncthreads();
  float vs = 0.f;
#pragma unroll
  for (int i = 0; i < 4; ++i) { float d = v[i] - mu; vs += d * d; }
  red[tid] = vs; __syncthreads();
  for (int off = 128; off; off >>= 1) { if (tid < off) red[tid] += red[tid + off]; __syncthreads(); }
  float inv = rsqrtf(red[0] * (1.f / DMODEL) + 1e-5f);
#pragma unroll
  for (int i = 0; i < 4; ++i) {
    int c = tid + i * 256;
    out[(long)row * DMODEL + c] = (bf16_t)((v[i] - mu) * inv * g[c] + beta[c]);
  }
}

// -------------------------------- GEMM -------------------------------------
// C[M,N] = A[M,K] (bf16 row-major) * Bt[N,K]^T (bf16) + bias, epilogue EPI.
// Block tile 128x256x32, 8 waves in 2x4; wave tile 64x64 = 4x4 WMMA tiles.
// LDS double-buffered; async DMA for tile t+1 overlaps WMMA on tile t.
#define BM 128
#define BN 256
#define BK 32
#define LDT 40   // 32 + 8 pad (80B rows -> conflict-free 16B fragment loads)

#define EPI_RESID_F32 0   // Cf = val + resid   (out-proj, ff2)
#define EPI_RELU_BF16 1   // Cb = relu(val)     (ff1)
#define EPI_QKV       2   // scatter q(*0.125)/k/vT per head

template <int EPI>
__global__ __launch_bounds__(256) void gemm_bf16_kernel(
    const bf16_t* __restrict__ A, const bf16_t* __restrict__ Bt,
    const float* __restrict__ bias, const float* __restrict__ resid,
    float* __restrict__ Cf, bf16_t* __restrict__ Cb,
    bf16_t* __restrict__ qout, bf16_t* __restrict__ kout,
    bf16_t* __restrict__ vtout, int M, int N, int K) {
  __shared__ __align__(16) bf16_t Asm[2][BM * LDT];   // 2 x 10 KB
  __shared__ __align__(16) bf16_t Bsm[2][BN * LDT];   // 2 x 20 KB
  int tid = threadIdx.x, lane = tid & 31, wid = tid >> 5;
  int bm = blockIdx.y * BM, bn = blockIdx.x * BN;
  int wm = (wid & 1) * 64, wn = (wid >> 1) * 64;

  v8f acc[4][4];
#pragma unroll
  for (int im = 0; im < 4; ++im)
#pragma unroll
    for (int jn = 0; jn < 4; ++jn)
#pragma unroll
      for (int e = 0; e < 8; ++e) acc[im][jn][e] = 0.f;

  auto stage = [&](int buf, int k0) {
#pragma unroll
    for (int c = 0; c < 2; ++c) {               // A: 512 x 16B chunks
      int ch = tid + c * 256;
      int r = ch >> 2, col = (ch & 3) * 8;
      g2l16(&Asm[buf][r * LDT + col], &A[(long)(bm + r) * K + k0 + col]);
    }
#pragma unroll
    for (int c = 0; c < 4; ++c) {               // B: 1024 x 16B chunks
      int ch = tid + c * 256;
      int r = ch >> 2, col = (ch & 3) * 8;
      g2l16(&Bsm[buf][r * LDT + col], &Bt[(long)(bn + r) * K + k0 + col]);
    }
  };

  int nchunks = K / BK;
  stage(0, 0);
  async_fence();
  __syncthreads();

  for (int t = 0; t < nchunks; ++t) {
    int cur = t & 1;
    if (t + 1 < nchunks) stage(cur ^ 1, (t + 1) * BK);
    if (t + 2 < nchunks) {                      // L2 prefetch two tiles ahead
      __builtin_prefetch(&A[(long)(bm + (tid >> 1)) * K + (t + 2) * BK], 0, 1);
      __builtin_prefetch(&Bt[(long)(bn + tid) * K + (t + 2) * BK], 0, 1);
    }
    v16bf af[4];
#pragma unroll
    for (int im = 0; im < 4; ++im)
      af[im] = load_fragA(&Asm[cur][(wm + im * 16) * LDT], LDT, lane);
#pragma unroll
    for (int jn = 0; jn < 4; ++jn) {
      v16bf bfg = load_fragB(&Bsm[cur][(wn + jn * 16) * LDT], LDT, lane);
#pragma unroll
      for (int im = 0; im < 4; ++im)
        acc[im][jn] = wmma_bf16(af[im], bfg, acc[im][jn]);
    }
    async_fence();
    __syncthreads();
  }

  // Epilogue. C layout: v8f elem i -> row i + 8*(lane>>4), col = lane&15.
  int ln = lane & 15, lh = lane >> 4;
#pragma unroll
  for (int im = 0; im < 4; ++im)
#pragma unroll
    for (int jn = 0; jn < 4; ++jn) {
      int c = bn + wn + jn * 16 + ln;
      float bv = bias[c];
#pragma unroll
      for (int i = 0; i < 8; ++i) {
        int r = bm + wm + im * 16 + i + 8 * lh;
        float val = acc[im][jn][i] + bv;
        if (EPI == EPI_RESID_F32) {
          Cf[(long)r * N + c] = val + resid[(long)r * N + c];
        } else if (EPI == EPI_RELU_BF16) {
          Cb[(long)r * N + c] = (bf16_t)fmaxf(val, 0.f);
        } else {  // EPI_QKV: c in [0,3D); split heads, scale q, transpose v
          int part = c >> 10, cc = c & 1023;
          int hh = cc >> 6, dd = cc & 63;
          int bb = r >> 11, ss = r & 2047;
          long bh = bb * NHEADS + hh;
          if (part == 0)      qout[(bh * SEQ + ss) * DK + dd] = (bf16_t)(val * 0.125f);
          else if (part == 1) kout[(bh * SEQ + ss) * DK + dd] = (bf16_t)val;
          else                vtout[(bh * DK + dd) * SEQ + ss] = (bf16_t)val;
        }
      }
    }
}

// -------------------------------- attention --------------------------------
// Flash-style: grid = (S/128, B*H). 8 waves; wave w owns q rows [w*16, w*16+16).
// Loops 128-key chunks: QK^T (8 WMMA tiles) -> running softmax -> P@V (LDS
// transpose of P via per-wave scratch; V staged as V^T rows in LDS).
#define LKD 72    // 64 + 8 pad (Q, K, P rows)
#define LVT 136   // 128 + 8 pad (V^T rows)

__global__ __launch_bounds__(256) void attn_kernel(
    const bf16_t* __restrict__ q, const bf16_t* __restrict__ k,
    const bf16_t* __restrict__ vT, bf16_t* __restrict__ o) {
  __shared__ __align__(16) bf16_t Ksm[128 * LKD];   // keys [key][d]
  __shared__ __align__(16) bf16_t Vsm[64 * LVT];    // V^T  [d][key]
  __shared__ __align__(16) bf16_t Qsm[128 * LKD];   // Q tile; reused as P scratch

  int tid = threadIdx.x, lane = tid & 31, wid = tid >> 5;
  int bh = blockIdx.y;
  int q0 = blockIdx.x * 128;
  const bf16_t* qbase  = q  + ((long)bh * SEQ + q0) * DK;
  const bf16_t* kbase  = k  + (long)bh * SEQ * DK;
  const bf16_t* vtbase = vT + (long)bh * DK * SEQ;

  // stage Q [128][64]
#pragma unroll
  for (int c = 0; c < 4; ++c) {
    int ch = tid + c * 256;            // 1024 x 16B chunks
    int r = ch >> 3, col = (ch & 7) * 8;
    g2l16(&Qsm[r * LKD + col], &qbase[(long)r * DK + col]);
  }
  async_fence();
  __syncthreads();
  v16bf qf[2];
#pragma unroll
  for (int t = 0; t < 2; ++t) qf[t] = load_fragA(&Qsm[(wid * 16) * LKD + t * 32], LKD, lane);
  __syncthreads();                     // Qsm now reusable as P scratch
  bf16_t* Pw = &Qsm[wid * 16 * LKD];   // per-wave 16x64(+pad) P region

  float m_run[8], l_run[8];
  v8f oacc[4];
#pragma unroll
  for (int i = 0; i < 8; ++i) { m_run[i] = -1e30f; l_run[i] = 0.f; }
#pragma unroll
  for (int jd = 0; jd < 4; ++jd)
#pragma unroll
    for (int e = 0; e < 8; ++e) oacc[jd][e] = 0.f;

  int ln = lane & 15, lh = lane >> 4;

  for (int kc = 0; kc < SEQ; kc += 128) {
    __syncthreads();
#pragma unroll
    for (int c = 0; c < 4; ++c) {      // K chunk [128][64]
      int ch = tid + c * 256;
      int r = ch >> 3, col = (ch & 7) * 8;
      g2l16(&Ksm[r * LKD + col], &kbase[(long)(kc + r) * DK + col]);
    }
#pragma unroll
    for (int c = 0; c < 4; ++c) {      // V^T chunk [64][128]
      int ch = tid + c * 256;
      int r = ch >> 4, col = (ch & 15) * 8;
      g2l16(&Vsm[r * LVT + col], &vtbase[(long)r * SEQ + kc + col]);
    }
    async_fence();
    __syncthreads();

    // S = Q * K^T : q pre-scaled by 1/sqrt(Dk)
    v8f sacc[8];
#pragma unroll
    for (int j = 0; j < 8; ++j) {
#pragma unroll
      for (int e = 0; e < 8; ++e) sacc[j][e] = 0.f;
#pragma unroll
      for (int t = 0; t < 2; ++t) {
        v16bf kf = load_fragB(&Ksm[(j * 16) * LKD + t * 32], LKD, lane);
        sacc[j] = wmma_bf16(qf[t], kf, sacc[j]);
      }
    }

    // running softmax per row (row values live across a 16-lane half-group)
#pragma unroll
    for (int i = 0; i < 8; ++i) {
      float mx = sacc[0][i];
#pragma unroll
      for (int j = 1; j < 8; ++j) mx = fmaxf(mx, sacc[j][i]);
#pragma unroll
      for (int msk = 1; msk < 16; msk <<= 1) mx = fmaxf(mx, __shfl_xor(mx, msk, 32));
      float mn = fmaxf(m_run[i], mx);
      float alpha = __expf(m_run[i] - mn);
      float ps = 0.f;
#pragma unroll
      for (int j = 0; j < 8; ++j) {
        float p = __expf(sacc[j][i] - mn);
        sacc[j][i] = p; ps += p;
      }
#pragma unroll
      for (int msk = 1; msk < 16; msk <<= 1) ps += __shfl_xor(ps, msk, 32);
      l_run[i] = l_run[i] * alpha + ps;
      m_run[i] = mn;
#pragma unroll
      for (int jd = 0; jd < 4; ++jd) oacc[jd][i] *= alpha;
    }

    // O += P * V, in two 64-key halves (per-wave LDS transpose of P; DS ops
    // from the same wave are processed in order, so no barrier needed here)
#pragma unroll
    for (int h2 = 0; h2 < 2; ++h2) {
#pragma unroll
      for (int jj = 0; jj < 4; ++jj)
#pragma unroll
        for (int i = 0; i < 8; ++i)
          Pw[(i + 8 * lh) * LKD + jj * 16 + ln] = (bf16_t)sacc[h2 * 4 + jj][i];
      v16bf pf[2];
#pragma unroll
      for (int t = 0; t < 2; ++t) pf[t] = load_fragA(&Pw[t * 32], LKD, lane);
#pragma unroll
      for (int jd = 0; jd < 4; ++jd)
#pragma unroll
        for (int t = 0; t < 2; ++t) {
          v16bf vf = load_fragB(&Vsm[(jd * 16) * LVT + h2 * 64 + t * 32], LVT, lane);
          oacc[jd] = wmma_bf16(pf[t], vf, oacc[jd]);
        }
    }
  }

  // normalize and write O into head-merged [B,S,D] bf16 layout
  int bb = bh >> 4, hh = bh & 15;
#pragma unroll
  for (int i = 0; i < 8; ++i) {
    float inv = 1.f / l_run[i];
    int srow = q0 + wid * 16 + i + 8 * lh;
#pragma unroll
    for (int jd = 0; jd < 4; ++jd) {
      long idx = ((long)bb * SEQ + srow) * DMODEL + hh * DK + jd * 16 + ln;
      o[idx] = (bf16_t)(oacc[jd][i] * inv);
    }
  }
}

// -------------------------------- launch -----------------------------------
extern "C" void kernel_launch(void* const* d_in, const int* in_sizes, int n_in,
                              void* d_out, int out_size, void* d_ws, size_t ws_size,
                              hipStream_t stream) {
  (void)in_sizes; (void)n_in; (void)out_size; (void)ws_size;
  const float* x     = (const float*)d_in[0];
  const float* w_qkv = (const float*)d_in[1];
  const float* b_qkv = (const float*)d_in[2];
  const float* w_out = (const float*)d_in[3];
  const float* b_out = (const float*)d_in[4];
  const float* w_ff1 = (const float*)d_in[5];
  const float* b_ff1 = (const float*)d_in[6];
  const float* w_ff2 = (const float*)d_in[7];
  const float* b_ff2 = (const float*)d_in[8];
  const float* g1    = (const float*)d_in[9];
  const float* be1   = (const float*)d_in[10];
  const float* g2    = (const float*)d_in[11];
  const float* be2   = (const float*)d_in[12];
  float* out = (float*)d_out;

  char* p = (char*)d_ws;
  auto carve = [&](size_t bytes) -> char* {
    char* r = p; p += (bytes + 255) & ~(size_t)255; return r;
  };
  bf16_t* wqkvT = (bf16_t*)carve((size_t)3 * DMODEL * DMODEL * 2);
  bf16_t* woutT = (bf16_t*)carve((size_t)DMODEL * DMODEL * 2);
  bf16_t* wff1T = (bf16_t*)carve((size_t)DFF * DMODEL * 2);
  bf16_t* wff2T = (bf16_t*)carve((size_t)DMODEL * DFF * 2);
  bf16_t* hb    = (bf16_t*)carve((size_t)ROWS * DMODEL * 2);     // LN1 out
  bf16_t* qb    = (bf16_t*)carve((size_t)ROWS * DMODEL * 2);     // [BH,S,Dk]
  bf16_t* kb    = (bf16_t*)carve((size_t)ROWS * DMODEL * 2);
  bf16_t* vtb   = (bf16_t*)carve((size_t)ROWS * DMODEL * 2);     // [BH,Dk,S]
  bf16_t* ob    = (bf16_t*)carve((size_t)ROWS * DMODEL * 2);     // attn out
  float*  x1    = (float*)carve((size_t)ROWS * DMODEL * 4);      // residual 1
  bf16_t* h2b   = (bf16_t*)carve((size_t)ROWS * DMODEL * 2);     // LN2 out
  bf16_t* hffb  = (bf16_t*)carve((size_t)ROWS * DFF * 2);        // relu(ff1)

  // one-time weight transpose + bf16 convert (W[K,N] -> Wt[N,K])
  cvt_transpose_kernel<<<(3 * DMODEL * DMODEL) / 256, 256, 0, stream>>>(w_qkv, wqkvT, DMODEL, 3 * DMODEL);
  cvt_transpose_kernel<<<(DMODEL * DMODEL) / 256, 256, 0, stream>>>(w_out, woutT, DMODEL, DMODEL);
  cvt_transpose_kernel<<<(DMODEL * DFF) / 256, 256, 0, stream>>>(w_ff1, wff1T, DMODEL, DFF);
  cvt_transpose_kernel<<<(DFF * DMODEL) / 256, 256, 0, stream>>>(w_ff2, wff2T, DFF, DMODEL);

  // h = LN1(x)
  layernorm_kernel<<<ROWS, 256, 0, stream>>>(x, g1, be1, hb);

  // qkv = h @ w_qkv + b_qkv  -> scatter q(*1/8)/k/vT
  gemm_bf16_kernel<EPI_QKV><<<dim3((3 * DMODEL) / BN, ROWS / BM), 256, 0, stream>>>(
      hb, wqkvT, b_qkv, nullptr, nullptr, nullptr, qb, kb, vtb, ROWS, 3 * DMODEL, DMODEL);

  // o = softmax(q k^T) v   (flash-style)
  attn_kernel<<<dim3(SEQ / 128, BATCH * NHEADS), 256, 0, stream>>>(qb, kb, vtb, ob);

  // x1 = x + (o @ w_out + b_out)
  gemm_bf16_kernel<EPI_RESID_F32><<<dim3(DMODEL / BN, ROWS / BM), 256, 0, stream>>>(
      ob, woutT, b_out, x, x1, nullptr, nullptr, nullptr, nullptr, ROWS, DMODEL, DMODEL);

  // h2 = LN2(x1)
  layernorm_kernel<<<ROWS, 256, 0, stream>>>(x1, g2, be2, h2b);

  // hff = relu(h2 @ w_ff1 + b_ff1)
  gemm_bf16_kernel<EPI_RELU_BF16><<<dim3(DFF / BN, ROWS / BM), 256, 0, stream>>>(
      h2b, wff1T, b_ff1, nullptr, nullptr, hffb, nullptr, nullptr, nullptr, ROWS, DFF, DMODEL);

  // out = x1 + (hff @ w_ff2 + b_ff2)
  gemm_bf16_kernel<EPI_RESID_F32><<<dim3(DMODEL / BN, ROWS / BM), 256, 0, stream>>>(
      hffb, wff2T, b_ff2, x1, out, nullptr, nullptr, nullptr, nullptr, ROWS, DMODEL, DFF);
}
